// MTPHead_57878979281267
// MI455X (gfx1250) — compile-verified, matrix operand order checked by reading, not run on hardware
//
#include <hip/hip_runtime.h>
#include <hip/hip_bf16.h>
#include <math.h>

typedef __bf16 bf16;
typedef __attribute__((ext_vector_type(8)))  __bf16 v8bf;
typedef __attribute__((ext_vector_type(16))) __bf16 v16bf;
typedef __attribute__((ext_vector_type(8)))  float  v8f;

#define B_   4
#define L_   2048
#define H_   1024
#define NQ_  8
#define NKV_ 2
#define D_   256
#define I_   3584
#define V_   16384
#define NT_  (B_*L_)          // 8192 tokens
#define EPS_ 1e-6f

// ---------------------------------------------------------------------------
// CDNA5 async global->LDS path (ASYNCcnt-tracked). The clang-22 builtin takes
// (v4i global*, v4i lds*, imm offset, imm cpol) per the probe diagnostic.
// Guarded so the build falls back to a register-staged pipeline if absent.
// ---------------------------------------------------------------------------
#if defined(__has_builtin)
#if __has_builtin(__builtin_amdgcn_global_load_async_to_lds_b128)
#define HAVE_ASYNC_LDS 1
#endif
#endif
#ifndef HAVE_ASYNC_LDS
#define HAVE_ASYNC_LDS 0
#endif

#if HAVE_ASYNC_LDS
typedef int vec4i_ __attribute__((vector_size(16)));
static __device__ __forceinline__ void async_ld_b128(const void* g, void* l) {
  __builtin_amdgcn_global_load_async_to_lds_b128(
      (__attribute__((address_space(1))) vec4i_*)g,
      (__attribute__((address_space(3))) vec4i_*)l, 0, 0);
}
static __device__ __forceinline__ void wait_async0() {
#if __has_builtin(__builtin_amdgcn_s_wait_asynccnt)
  __builtin_amdgcn_s_wait_asynccnt(0);
#else
  asm volatile("s_wait_asynccnt 0" ::: "memory");
#endif
}
#endif

// ---------------------------------------------------------------------------
// WMMA helpers (CDNA5 gfx1250, wave32).
// 16-bit A fragment map (ISA 7.12.2): lane m = lane&15, half = lane>>4,
//   elem i<8  -> k = half*8 + i
//   elem i>=8 -> k = 16 + half*8 + (i-8)
// B fragment mirrors A with n = lane&15 (B^T row-major reads).
// ---------------------------------------------------------------------------
static __device__ __forceinline__ v16bf mk16(v8bf lo, v8bf hi) {
  v16bf r;
#pragma unroll
  for (int i = 0; i < 8; i++) { r[i] = lo[i]; r[8 + i] = hi[i]; }
  return r;
}
static __device__ __forceinline__ v8f wmma_bf16(v16bf a, v16bf b, v8f c) {
  return __builtin_amdgcn_wmma_f32_16x16x32_bf16(false, a, false, b,
                                                 (short)0, c, false, false);
}

// ---------------------------------------------------------------------------
// GEMM: C[M,N] (f32) = A[M,K] (bf16 row-major) * W[N,K]^T (bf16 row-major)
// N, K compile-time: C-store addressing folds to base + immediate offsets.
// Block tile 128x128x32, 256 threads = 8 waves (2x4), each wave 64x32.
// Double-buffered LDS, one barrier per K-stage; async global->LDS if present.
// ---------------------------------------------------------------------------
template <int N, int K, bool NTSTORE>
__global__ __launch_bounds__(256) void gemm_t(
    const bf16* __restrict__ A, const bf16* __restrict__ W,
    float* __restrict__ C)
{
  __shared__ __align__(16) bf16 As[2][128 * 40];   // 32 + 8 pad halves per row
  __shared__ __align__(16) bf16 Ws[2][128 * 40];
  const int tid  = threadIdx.x;
  const int wave = tid >> 5, lane = tid & 31;
  const int hf   = lane >> 4, mrow = lane & 15;
  const int wm   = wave >> 2, wn = wave & 3;       // 2 x 4 wave grid
  const int row0 = blockIdx.x * 128;
  const int col0 = blockIdx.y * 128;

  // Per-thread staging coords: 512 16B chunks per 128x32 tile, 2 per thread
  // (rows r and r+64, same 8-half column group c).
  const int r = tid >> 2, c = tid & 3;
  const bf16* pA = A + (size_t)(row0 + r) * K + c * 8;
  const bf16* pW = W + (size_t)(col0 + r) * K + c * 8;
  const int ldso = r * 40 + c * 8;                 // +64*40 for second chunk

  v8f acc[4][2] = {};
  constexpr int S = K / 32;

  auto compute_stage = [&](const bf16* as, const bf16* ws) {
    v16bf af[4], wf[2];
#pragma unroll
    for (int mt = 0; mt < 4; mt++) {
      int base = (wm * 64 + mt * 16 + mrow) * 40 + hf * 8;
      af[mt] = mk16(*(const v8bf*)&as[base], *(const v8bf*)&as[base + 16]);
    }
#pragma unroll
    for (int nt = 0; nt < 2; nt++) {
      int base = (wn * 32 + nt * 16 + mrow) * 40 + hf * 8;
      wf[nt] = mk16(*(const v8bf*)&ws[base], *(const v8bf*)&ws[base + 16]);
    }
#pragma unroll
    for (int mt = 0; mt < 4; mt++)
#pragma unroll
      for (int nt = 0; nt < 2; nt++)
        acc[mt][nt] = wmma_bf16(af[mt], wf[nt], acc[mt][nt]);
  };

#if HAVE_ASYNC_LDS
  // ---- async pipeline: prefetch stage s+1 while WMMAs consume stage s ----
  async_ld_b128(pA,          &As[0][ldso]);
  async_ld_b128(pA + 64 * K, &As[0][ldso + 64 * 40]);
  async_ld_b128(pW,          &Ws[0][ldso]);
  async_ld_b128(pW + 64 * K, &Ws[0][ldso + 64 * 40]);
#pragma unroll 1
  for (int s = 0; s < S; ++s) {
    const int buf = s & 1;
    wait_async0();                 // this wave's share of stage s has landed
    __syncthreads();               // everyone's share has landed
    if (s + 1 < S) {
      const bf16* nA = pA + (size_t)(s + 1) * 32;
      const bf16* nW = pW + (size_t)(s + 1) * 32;
      async_ld_b128(nA,          &As[1 - buf][ldso]);
      async_ld_b128(nA + 64 * K, &As[1 - buf][ldso + 64 * 40]);
      async_ld_b128(nW,          &Ws[1 - buf][ldso]);
      async_ld_b128(nW + 64 * K, &Ws[1 - buf][ldso + 64 * 40]);
    }
    compute_stage(As[buf], Ws[buf]);
  }
#else
  // ---- fallback: register-staged double buffer, one barrier per stage ----
  uint4 a0 = *(const uint4*)pA;
  uint4 a1 = *(const uint4*)(pA + 64 * K);
  uint4 w0 = *(const uint4*)pW;
  uint4 w1 = *(const uint4*)(pW + 64 * K);
#pragma unroll 1
  for (int s = 0; s < S; ++s) {
    const int buf = s & 1;
    *(uint4*)&As[buf][ldso]           = a0;
    *(uint4*)&As[buf][ldso + 64 * 40] = a1;
    *(uint4*)&Ws[buf][ldso]           = w0;
    *(uint4*)&Ws[buf][ldso + 64 * 40] = w1;
    __syncthreads();
    if (s + 1 < S) {                 // global loads overlap with WMMAs below
      const bf16* nA = pA + (size_t)(s + 1) * 32;
      const bf16* nW = pW + (size_t)(s + 1) * 32;
      a0 = *(const uint4*)nA;  a1 = *(const uint4*)(nA + 64 * K);
      w0 = *(const uint4*)nW;  w1 = *(const uint4*)(nW + 64 * K);
    }
    compute_stage(As[buf], Ws[buf]);
  }
#endif

  // C fragment: VGPR v -> row m = v + 8*half, col n = lane&15.
  // With constant N the 64 stores become base + immediate offsets.
  float* cbase = C + (size_t)(row0 + wm * 64 + 8 * hf) * N
                   + col0 + wn * 32 + mrow;
#pragma unroll
  for (int mt = 0; mt < 4; mt++)
#pragma unroll
    for (int nt = 0; nt < 2; nt++)
#pragma unroll
      for (int v = 0; v < 8; v++) {
        float* dst = cbase + (size_t)(mt * 16 + v) * N + nt * 16;
        if (NTSTORE) __builtin_nontemporal_store(acc[mt][nt][v], dst);
        else         *dst = acc[mt][nt][v];
      }
}

// ---------------------------------------------------------------------------
// rmsnorm(emb)*we ++ rmsnorm(hid)*wh -> bf16 [NT, 2048]. One block per token.
// ---------------------------------------------------------------------------
__global__ __launch_bounds__(256) void rmsnorm_concat_kernel(
    const float* __restrict__ emb, const float* __restrict__ hid,
    const float* __restrict__ we,  const float* __restrict__ wh,
    bf16* __restrict__ out)
{
  __shared__ float red[256];
  int t = blockIdx.x, tid = threadIdx.x;
  float xe[4], xh[4], se = 0.f, sh = 0.f;
#pragma unroll
  for (int i = 0; i < 4; i++) {
    int j = tid + i * 256;
    xe[i] = emb[(size_t)t * H_ + j];
    xh[i] = hid[(size_t)t * H_ + j];
    se += xe[i] * xe[i]; sh += xh[i] * xh[i];
  }
  red[tid] = se; __syncthreads();
  for (int s = 128; s > 0; s >>= 1) { if (tid < s) red[tid] += red[tid + s]; __syncthreads(); }
  float rmse = rsqrtf(red[0] / (float)H_ + EPS_); __syncthreads();
  red[tid] = sh; __syncthreads();
  for (int s = 128; s > 0; s >>= 1) { if (tid < s) red[tid] += red[tid + s]; __syncthreads(); }
  float rmsh = rsqrtf(red[0] / (float)H_ + EPS_);
#pragma unroll
  for (int i = 0; i < 4; i++) {
    int j = tid + i * 256;
    out[(size_t)t * 2048 + j]        = (bf16)(xe[i] * rmse * we[j]);
    out[(size_t)t * 2048 + 1024 + j] = (bf16)(xh[i] * rmsh * wh[j]);
  }
}

// rmsnorm over H=1024, f32 in -> bf16 out. One block per token.
__global__ __launch_bounds__(256) void rmsnorm1024_kernel(
    const float* __restrict__ x, const float* __restrict__ w, bf16* __restrict__ y)
{
  __shared__ float red[256];
  int t = blockIdx.x, tid = threadIdx.x;
  float xr[4], ss = 0.f;
#pragma unroll
  for (int i = 0; i < 4; i++) {
    int j = tid + i * 256;
    xr[i] = x[(size_t)t * H_ + j];
    ss += xr[i] * xr[i];
  }
  red[tid] = ss; __syncthreads();
  for (int s = 128; s > 0; s >>= 1) { if (tid < s) red[tid] += red[tid + s]; __syncthreads(); }
  float rms = rsqrtf(red[0] / (float)H_ + EPS_);
#pragma unroll
  for (int i = 0; i < 4; i++) {
    int j = tid + i * 256;
    y[(size_t)t * H_ + j] = (bf16)(xr[i] * rms * w[j]);
  }
}

// ---------------------------------------------------------------------------
// Per-head rmsnorm + RoPE (theta=1e7, first 64 dims), optional output scale.
// grid = B*L*nh blocks, 256 threads (one per D element).
// dst layout: ((b*nh + h)*L + l)*256 + d  (head-major for attention tiles).
// ---------------------------------------------------------------------------
__global__ __launch_bounds__(256) void qk_rope_kernel(
    const float* __restrict__ src, const float* __restrict__ w,
    bf16* __restrict__ dst, int nh, int tstride, int hstride, float outscale)
{
  __shared__ float red[256];
  __shared__ float sh[256];
  int idx = blockIdx.x;
  int h = idx % nh, t = idx / nh;
  int b = t / L_,  l = t % L_;
  int d = threadIdx.x;
  float x = src[(size_t)t * tstride + h * hstride + d];
  red[d] = x * x; __syncthreads();
  for (int s = 128; s > 0; s >>= 1) { if (d < s) red[d] += red[d + s]; __syncthreads(); }
  float rms = rsqrtf(red[0] / (float)D_ + EPS_);
  float xn = x * rms * w[d];
  sh[d] = xn; __syncthreads();
  size_t ob = ((size_t)(b * nh + h) * L_ + l) * D_;
  if (d < 32) {                       // 32 rotary pairs
    float inv = __powf(1.0e7f, -(float)d / 32.0f);
    float f = (float)l * inv;
    float cc = __cosf(f), ss = __sinf(f);
    float x0 = sh[2 * d], x1 = sh[2 * d + 1];
    dst[ob + 2 * d]     = (bf16)((x0 * cc - x1 * ss) * outscale);
    dst[ob + 2 * d + 1] = (bf16)((x0 * ss + x1 * cc) * outscale);
  } else if (d >= 64) {
    dst[ob + d] = (bf16)(xn * outscale);
  }
}

// V: (b,l,hkv,d) f32 -> (b,hkv,d,l) bf16 so flash B-fragments are contiguous.
__global__ void v_transpose_kernel(const float* __restrict__ v, bf16* __restrict__ out)
{
  long long i = (long long)blockIdx.x * blockDim.x + threadIdx.x;
  long long n = (long long)B_ * NKV_ * D_ * L_;
  if (i >= n) return;
  int l = (int)(i % L_);
  int d = (int)((i / L_) % D_);
  int h = (int)((i / ((long long)L_ * D_)) % NKV_);
  int b = (int)(i / ((long long)L_ * D_ * NKV_));
  int t = b * L_ + l;
  out[i] = (bf16)v[(size_t)t * (NKV_ * D_) + h * D_ + d];
}

// ---------------------------------------------------------------------------
// Flash attention. Block = (b, h, 16 q rows), 128 threads = 4 waves.
// Wave w owns D columns [w*64, w*64+64). Q pre-scaled by 1/sqrt(D).
// 32-wide KV tiles: P is a full-K (16x32) A fragment -> no zero padding.
// ---------------------------------------------------------------------------
__global__ __launch_bounds__(128) void attention_kernel(
    const bf16* __restrict__ Q, const bf16* __restrict__ Kt,
    const bf16* __restrict__ Vt, float* __restrict__ O)
{
  __shared__ __align__(16) bf16 Qs[16 * 256];      //  8 KB
  __shared__ __align__(16) bf16 Ks[32 * 256];      // 16 KB
  __shared__ __align__(16) bf16 Vs[256 * 32];      // 16 KB, [d][kv]
  __shared__ __align__(16) bf16 Ps[4][16 * 32];    //  4 KB, per-wave P tile
  const int tid  = threadIdx.x;
  const int wave = tid >> 5, lane = tid & 31;
  const int hf   = lane >> 4, mrow = lane & 15, ncol = lane & 15;
  const int qb   = blockIdx.x * 16;
  const int h    = blockIdx.y;
  const int b    = blockIdx.z;
  const int hkv  = h >> 2;                         // NQ/NKV = 4

  const size_t qoff = ((size_t)(b * NQ_ + h) * L_ + qb) * D_;
  const size_t kof  = ((size_t)(b * NKV_ + hkv) * L_) * D_;
  const size_t vof  = ((size_t)(b * NKV_ + hkv) * D_) * L_;

#pragma unroll
  for (int it = 0; it < 4; it++) {                 // Q tile: 512 16B chunks
    int idx = tid + it * 128;
    int rr = idx >> 5, cc = idx & 31;
    *reinterpret_cast<uint4*>(&Qs[rr * 256 + cc * 8]) =
        *reinterpret_cast<const uint4*>(Q + qoff + (size_t)rr * D_ + cc * 8);
  }

  v8f oacc[4] = {};
  float mprev[8], lrun[8];
#pragma unroll
  for (int v = 0; v < 8; v++) { mprev[v] = -1e30f; lrun[v] = 0.f; }

  // Largest kb is the biggest multiple of 32 below qb+16, so kb+31 <= L-1
  // always (no OOB K/V rows; masked columns stay finite -> p exactly 0).
  for (int kb = 0; kb < qb + 16; kb += 32) {
    __syncthreads();
#pragma unroll
    for (int it = 0; it < 8; it++) {               // K tile 32x256: 1024 chunks
      int idx = tid + it * 128;
      int rr = idx >> 5, cc = idx & 31;
      *reinterpret_cast<uint4*>(&Ks[rr * 256 + cc * 8]) =
          *reinterpret_cast<const uint4*>(Kt + kof + (size_t)(kb + rr) * D_ + cc * 8);
    }
#pragma unroll
    for (int it = 0; it < 8; it++) {               // V tile 256x32 (d-major)
      int idx = tid + it * 128;
      int d = idx >> 2, cc = idx & 3;
      *reinterpret_cast<uint4*>(&Vs[d * 32 + cc * 8]) =
          *reinterpret_cast<const uint4*>(Vt + vof + (size_t)d * L_ + kb + cc * 8);
    }
    __syncthreads();

    // S = Q K^T over D=256, two 16x16 column tiles (redundant per wave)
    v8f s0 = {}, s1 = {};
#pragma unroll
    for (int ko = 0; ko < 8; ko++) {
      int ab = mrow * 256 + ko * 32 + hf * 8;
      v16bf a = mk16(*(const v8bf*)&Qs[ab], *(const v8bf*)&Qs[ab + 16]);
      int b0 = ncol * 256 + ko * 32 + hf * 8;
      v16bf k0 = mk16(*(const v8bf*)&Ks[b0], *(const v8bf*)&Ks[b0 + 16]);
      s0 = wmma_bf16(a, k0, s0);
      int b1 = (16 + ncol) * 256 + ko * 32 + hf * 8;
      v16bf k1 = mk16(*(const v8bf*)&Ks[b1], *(const v8bf*)&Ks[b1 + 16]);
      s1 = wmma_bf16(a, k1, s1);
    }
    if (kb + 31 > qb) {                            // causal tail tiles
#pragma unroll
      for (int v = 0; v < 8; v++) {
        int m = qb + v + 8 * hf;
        if (kb + ncol      > m) s0[v] = -1e30f;
        if (kb + 16 + ncol > m) s1[v] = -1e30f;
      }
    }
    // online softmax over 32 columns (rows m = v + 8*half across half-wave)
    float p0[8], p1[8], alpha[8];
#pragma unroll
    for (int v = 0; v < 8; v++) {
      float rm = fmaxf(s0[v], s1[v]);
#pragma unroll
      for (int msk = 8; msk >= 1; msk >>= 1) rm = fmaxf(rm, __shfl_xor(rm, msk, 32));
      float mnew = fmaxf(mprev[v], rm);
      alpha[v] = __expf(mprev[v] - mnew);
      p0[v] = __expf(s0[v] - mnew);
      p1[v] = __expf(s1[v] - mnew);
      float rs = p0[v] + p1[v];
#pragma unroll
      for (int msk = 8; msk >= 1; msk >>= 1) rs += __shfl_xor(rs, msk, 32);
      lrun[v] = lrun[v] * alpha[v] + rs;
      mprev[v] = mnew;
    }
#pragma unroll
    for (int v = 0; v < 8; v++) {
#pragma unroll
      for (int nt = 0; nt < 4; nt++) oacc[nt][v] *= alpha[v];
      int prow = (v + 8 * hf) * 32;
      Ps[wave][prow + ncol]      = (bf16)p0[v];
      Ps[wave][prow + 16 + ncol] = (bf16)p1[v];
    }
    // same-wave LDS RAW across lanes: DS pipe is in-order; fence the compiler
    asm volatile("s_wait_dscnt 0" ::: "memory");

    // O += P (16x32, full K) @ V (32 kv x 64 d slice)
    int pb = mrow * 32 + hf * 8;
    v16bf pa = mk16(*(const v8bf*)&Ps[wave][pb], *(const v8bf*)&Ps[wave][pb + 16]);
#pragma unroll
    for (int nt = 0; nt < 4; nt++) {
      int dcol = wave * 64 + nt * 16 + ncol;
      int vb0 = dcol * 32 + hf * 8;
      v16bf vb = mk16(*(const v8bf*)&Vs[vb0], *(const v8bf*)&Vs[vb0 + 16]);
      oacc[nt] = wmma_bf16(pa, vb, oacc[nt]);
    }
  }

  float inv[8];
#pragma unroll
  for (int v = 0; v < 8; v++) inv[v] = 1.0f / lrun[v];
#pragma unroll
  for (int nt = 0; nt < 4; nt++) {
    int dcol = wave * 64 + nt * 16 + ncol;
#pragma unroll
    for (int v = 0; v < 8; v++) {
      int q = qb + v + 8 * hf;
      O[((size_t)(b * L_ + q) * NQ_ + h) * D_ + dcol] = oacc[nt][v] * inv[v];
    }
  }
}

// ---------------------------------------------------------------------------
// Elementwise kernels
// ---------------------------------------------------------------------------
__global__ void f32_to_bf16_kernel(const float* __restrict__ x, bf16* __restrict__ y, long long n)
{
  long long i = (long long)blockIdx.x * blockDim.x + threadIdx.x;
  if (i < n) y[i] = (bf16)x[i];
}

__global__ void gate_apply_kernel(const float* __restrict__ attn,
                                  const float* __restrict__ qg, bf16* __restrict__ out)
{
  long long i = (long long)blockIdx.x * blockDim.x + threadIdx.x;
  if (i >= (long long)NT_ * 2048) return;
  int t = (int)(i / 2048), c = (int)(i % 2048);
  int h = c >> 8, d = c & 255;
  float g = qg[(size_t)t * 4096 + h * 512 + 256 + d];   // gate slice of QG proj
  float sg = 1.0f / (1.0f + __expf(-g));
  out[i] = (bf16)(attn[i] * sg);
}

__global__ void residual_add_kernel(const float* __restrict__ a,
                                    const float* __restrict__ b,
                                    float* __restrict__ y, long long n)
{
  long long i = (long long)blockIdx.x * blockDim.x + threadIdx.x;
  if (i < n) y[i] = a[i] + b[i];
}

__global__ void silu_mul_kernel(const float* __restrict__ g,
                                const float* __restrict__ u,
                                bf16* __restrict__ y, long long n)
{
  long long i = (long long)blockIdx.x * blockDim.x + threadIdx.x;
  if (i < n) {
    float gv = g[i];
    y[i] = (bf16)((gv / (1.0f + __expf(-gv))) * u[i]);
  }
}

// ---------------------------------------------------------------------------
// Host-side orchestration
// ---------------------------------------------------------------------------
static inline int ceil_div_i(long long a, int b) { return (int)((a + b - 1) / b); }

extern "C" void kernel_launch(void* const* d_in, const int* in_sizes, int n_in,
                              void* d_out, int out_size, void* d_ws, size_t ws_size,
                              hipStream_t stream)
{
  (void)in_sizes; (void)n_in; (void)out_size; (void)ws_size;
  const float* emb      = (const float*)d_in[0];
  const float* hid      = (const float*)d_in[1];
  const float* pre_e_w  = (const float*)d_in[2];
  const float* pre_h_w  = (const float*)d_in[3];
  const float* fc_w     = (const float*)d_in[4];
  const float* in_ln_w  = (const float*)d_in[5];
  const float* q_w      = (const float*)d_in[6];
  const float* k_w      = (const float*)d_in[7];
  const float* v_w      = (const float*)d_in[8];
  const float* o_w      = (const float*)d_in[9];
  const float* qn_w     = (const float*)d_in[10];
  const float* kn_w     = (const float*)d_in[11];
  const float* post_ln  = (const float*)d_in[12];
  const float* gate_w   = (const float*)d_in[13];
  const float* up_w     = (const float*)d_in[14];
  const float* down_w   = (const float*)d_in[15];
  const float* norm_w   = (const float*)d_in[16];
  const float* lm_w     = (const float*)d_in[17];
  float* out            = (float*)d_out;

  // ---- workspace carve (every region fully written before read each call) ----
  char* p = (char*)d_ws;
  auto alloc = [&](size_t bytes) -> void* {
    void* r = (void*)p; p += (bytes + 255) & ~(size_t)255; return r;
  };
  bf16* fcw  = (bf16*)alloc((size_t)H_ * 2 * H_ * 2);
  bf16* qw   = (bf16*)alloc((size_t)2 * NQ_ * D_ * H_ * 2);
  bf16* kw   = (bf16*)alloc((size_t)NKV_ * D_ * H_ * 2);
  bf16* vw   = (bf16*)alloc((size_t)NKV_ * D_ * H_ * 2);
  bf16* ow   = (bf16*)alloc((size_t)H_ * NQ_ * D_ * 2);
  bf16* gw   = (bf16*)alloc((size_t)I_ * H_ * 2);
  bf16* uw   = (bf16*)alloc((size_t)I_ * H_ * 2);
  bf16* dw   = (bf16*)alloc((size_t)H_ * I_ * 2);
  bf16* lmw  = (bf16*)alloc((size_t)V_ * H_ * 2);

  bf16*  xcat     = (bf16*) alloc((size_t)NT_ * 2048 * 2);
  float* fc_out   = (float*)alloc((size_t)NT_ * 1024 * 4);
  bf16*  xn       = (bf16*) alloc((size_t)NT_ * 1024 * 2);
  float* qg       = (float*)alloc((size_t)NT_ * 4096 * 4);
  float* kproj    = (float*)alloc((size_t)NT_ * 512 * 4);
  float* vproj    = (float*)alloc((size_t)NT_ * 512 * 4);
  bf16*  Qb       = (bf16*) alloc((size_t)B_ * NQ_ * L_ * D_ * 2);
  bf16*  Kb       = (bf16*) alloc((size_t)B_ * NKV_ * L_ * D_ * 2);
  bf16*  Vb       = (bf16*) alloc((size_t)B_ * NKV_ * L_ * D_ * 2);
  float* attn_out = (float*)alloc((size_t)NT_ * 2048 * 4);
  bf16*  attn_g   = (bf16*) alloc((size_t)NT_ * 2048 * 2);
  float* o_out    = (float*)alloc((size_t)NT_ * 1024 * 4);
  float* x2       = (float*)alloc((size_t)NT_ * 1024 * 4);
  bf16*  xn2      = (bf16*) alloc((size_t)NT_ * 1024 * 2);
  // Aliases (source regions are dead by the time these are written):
  float* g_out    = qg;               // NT*3584 f32 fits in NT*4096 f32
  float* u_out    = attn_out;         // fits attn_out+attn_g+o_out span
  bf16*  h_mlp    = xcat;             // NT*3584 bf16 fits xcat+fc_out span
  float* down_out = o_out;            // within dead attn span
  float* x3       = kproj;            // NT*1024 f32 fits kproj+vproj span
  bf16*  xn3      = Kb;               // NT*1024 bf16 fits Kb+Vb span

  // ---- 1) weights -> bf16 ----
  struct WCv { const float* s; bf16* d; long long n; } wc[9] = {
    { fc_w,   fcw, (long long)H_ * 2 * H_       },
    { q_w,    qw,  (long long)2 * NQ_ * D_ * H_ },
    { k_w,    kw,  (long long)NKV_ * D_ * H_    },
    { v_w,    vw,  (long long)NKV_ * D_ * H_    },
    { o_w,    ow,  (long long)H_ * NQ_ * D_     },
    { gate_w, gw,  (long long)I_ * H_           },
    { up_w,   uw,  (long long)I_ * H_           },
    { down_w, dw,  (long long)H_ * I_           },
    { lm_w,   lmw, (long long)V_ * H_           },
  };
  for (int i = 0; i < 9; i++)
    f32_to_bf16_kernel<<<ceil_div_i(wc[i].n, 256), 256, 0, stream>>>(wc[i].s, wc[i].d, wc[i].n);

  // ---- 2) concat of rmsnorms ----
  rmsnorm_concat_kernel<<<NT_, 256, 0, stream>>>(emb, hid, pre_e_w, pre_h_w, xcat);

  // ---- 3) fc projection (8192 x 1024 x 2048) ----
  gemm_t<1024, 2048, false><<<dim3(NT_ / 128, 1024 / 128), 256, 0, stream>>>(xcat, fcw, fc_out);

  // ---- 4) input layernorm ----
  rmsnorm1024_kernel<<<NT_, 256, 0, stream>>>(fc_out, in_ln_w, xn);

  // ---- 5) QKV projections ----
  gemm_t<4096, 1024, false><<<dim3(NT_ / 128, 4096 / 128), 256, 0, stream>>>(xn, qw, qg);
  gemm_t< 512, 1024, false><<<dim3(NT_ / 128,  512 / 128), 256, 0, stream>>>(xn, kw, kproj);
  gemm_t< 512, 1024, false><<<dim3(NT_ / 128,  512 / 128), 256, 0, stream>>>(xn, vw, vproj);

  // ---- 6) head norms + RoPE (1/sqrt(D)=1/16 folded into Q), V transpose ----
  qk_rope_kernel<<<B_ * L_ * NQ_, 256, 0, stream>>>(qg, qn_w, Qb, NQ_, 4096, 512, 0.0625f);
  qk_rope_kernel<<<B_ * L_ * NKV_, 256, 0, stream>>>(kproj, kn_w, Kb, NKV_, 512, 256, 1.0f);
  v_transpose_kernel<<<ceil_div_i((long long)B_ * NKV_ * D_ * L_, 256), 256, 0, stream>>>(vproj, Vb);

  // ---- 7) flash attention ----
  attention_kernel<<<dim3(L_ / 16, NQ_, B_), 128, 0, stream>>>(Qb, Kb, Vb, attn_out);

  // ---- 8) sigmoid gate, o-projection, residual ----
  gate_apply_kernel<<<ceil_div_i((long long)NT_ * 2048, 256), 256, 0, stream>>>(attn_out, qg, attn_g);
  gemm_t<1024, 2048, false><<<dim3(NT_ / 128, 1024 / 128), 256, 0, stream>>>(attn_g, ow, o_out);
  residual_add_kernel<<<ceil_div_i((long long)NT_ * 1024, 256), 256, 0, stream>>>(fc_out, o_out, x2, (long long)NT_ * 1024);

  // ---- 9) MLP ----
  rmsnorm1024_kernel<<<NT_, 256, 0, stream>>>(x2, post_ln, xn2);
  gemm_t<I_, 1024, false><<<dim3(NT_ / 128, I_ / 128), 256, 0, stream>>>(xn2, gw, g_out);
  gemm_t<I_, 1024, false><<<dim3(NT_ / 128, I_ / 128), 256, 0, stream>>>(xn2, uw, u_out);
  silu_mul_kernel<<<ceil_div_i((long long)NT_ * I_, 256), 256, 0, stream>>>(g_out, u_out, h_mlp, (long long)NT_ * I_);
  gemm_t<1024, I_, false><<<dim3(NT_ / 128, 1024 / 128), 256, 0, stream>>>(h_mlp, dw, down_out);
  residual_add_kernel<<<ceil_div_i((long long)NT_ * 1024, 256), 256, 0, stream>>>(x2, down_out, x3, (long long)NT_ * 1024);

  // ---- 10) final norm + lm head (8192 x 16384 x 1024) -> d_out ----
  rmsnorm1024_kernel<<<NT_, 256, 0, stream>>>(x3, norm_w, xn3);
  gemm_t<V_, 1024, true><<<dim3(NT_ / 128, V_ / 128), 256, 0, stream>>>(xn3, lmw, out);
}
